// DotProductAttention_20469814133283
// MI455X (gfx1250) — matrix operand order repaired and checked by reading, not compiled, not run on hardware
//
#include <hip/hip_runtime.h>
#include <hip/hip_bf16.h>

typedef unsigned short u16;
typedef unsigned int   u32;
typedef unsigned long long u64;
typedef __attribute__((ext_vector_type(16))) __bf16 v16bf;
typedef __attribute__((ext_vector_type(8)))  float  v8f;
typedef __attribute__((ext_vector_type(4)))  u32    v4u;
typedef __attribute__((ext_vector_type(8)))  int    v8i;
typedef __attribute__((ext_vector_type(4)))  int    v4i;

// Problem constants (B,Q,K,d) = (16,2048,2048,512)
#define BB   16
#define QQ   2048
#define KK   2048
#define DD   512
#define TILE 128      // q/k tile per workgroup
#define DC   64       // d-chunk staged in LDS
#define NCH  (DD / DC)
#define SAS  72       // LDS row stride in bf16 elems (64 + 8 pad -> 144B, 16B aligned)
#define SCALE 0.044194173824159216f   // 1/sqrt(512)
#define NEGMAX (-3.402823466e38f)

#if __has_builtin(__builtin_amdgcn_tensor_load_to_lds) && \
    __has_builtin(__builtin_amdgcn_s_wait_tensorcnt)
#define USE_TDM 1
#else
#define USE_TDM 0
#endif

union Frag { uint4 u[2]; v16bf v; };

__device__ __forceinline__ u32 pack_bf16(float a, float b) {
  u32 ua = __float_as_uint(a); ua = (ua + 0x7FFFu + ((ua >> 16) & 1u)) >> 16;
  u32 ub = __float_as_uint(b); ub = (ub + 0x7FFFu + ((ub >> 16) & 1u)) >> 16;
  return ua | (ub << 16);
}

// ---------------- K1: f32 -> bf16 convert (RNE), vectorized ----------------
__global__ __launch_bounds__(256)
void attn_cvt_bf16(const float* __restrict__ in, u16* __restrict__ out, int n4) {
  int i = blockIdx.x * blockDim.x + threadIdx.x;
  int stride = gridDim.x * blockDim.x;
  for (; i < n4; i += stride) {
    float4 f = ((const float4*)in)[i];
    uint2 p; p.x = pack_bf16(f.x, f.y); p.y = pack_bf16(f.z, f.w);
    ((uint2*)out)[i] = p;
  }
}

#if USE_TDM
// ------- TDM: DMA one 128x64 bf16 tile (row stride DD elems) into LDS ------
// D# per CDNA5 ISA ch.8: group0 = {count/type/addr}, group1 = dims/strides.
// pad_interval=4 (32 dwords = 64 bf16) + pad_amount=3 (4 dwords = 8 bf16)
// -> hardware writes the 144B padded row stride (SAS) for us.
__device__ __forceinline__ void tdm_load_tile(const u16* gsrc, u32 lds_addr) {
  u64 ga = (u64)(uintptr_t)gsrc;
  v4u g0;
  g0.x = 1u;                                             // count=1, user mode
  g0.y = lds_addr;                                       // LDS byte address
  g0.z = (u32)ga;                                        // global addr [31:0]
  g0.w = ((u32)(ga >> 32) & 0x01FFFFFFu) | (2u << 30);   // addr[56:32] | type=2
  v8i g1;
  g1[0] = (int)((1u << 16) | (1u << 20) | (4u << 22) | (3u << 25)); // 2B, pad on
  g1[1] = (int)((u32)(DD & 0xFFFF) << 16);               // tensor_dim0 = 512
  g1[2] = (int)((u32)(DD >> 16) | ((u32)(TILE & 0xFFFF) << 16)); // tensor_dim1=128
  g1[3] = (int)((u32)(TILE >> 16) | ((u32)DC << 16));    // tile_dim0 = 64
  g1[4] = (int)(TILE);                                   // tile_dim1 = 128, tile_dim2=0
  g1[5] = (int)(DD);                                     // tensor_dim0_stride = 512
  g1[6] = 0; g1[7] = 0;
  v4i gz = {0, 0, 0, 0};
#if defined(__clang_major__) && __clang_major__ >= 23
  v8i gz8 = {0, 0, 0, 0, 0, 0, 0, 0};
  __builtin_amdgcn_tensor_load_to_lds(g0, g1, gz, gz, gz8, 0);
#else
  __builtin_amdgcn_tensor_load_to_lds(g0, g1, gz, gz, 0);
#endif
}
#endif

// -------- fallback: cooperative synchronous 128x64 chunk load to LDS -------
__device__ __forceinline__ void load_chunk(u16* __restrict__ sA, u16* __restrict__ sB,
                                           const u16* __restrict__ ga,
                                           const u16* __restrict__ gb,
                                           int dc, int tid) {
  int rr = tid >> 3;            // 0..31
  int cc = (tid & 7) * 8;       // bf16 elem offset 0..56 (16B granular)
#pragma unroll
  for (int p = 0; p < 4; ++p) {
    int row = rr + p * 32;
    *(uint4*)&sA[row * SAS + cc] = *(const uint4*)(ga + (size_t)row * DD + dc * DC + cc);
    *(uint4*)&sB[row * SAS + cc] = *(const uint4*)(gb + (size_t)row * DD + dc * DC + cc);
  }
}

// -------- 16 WMMAs on one staged chunk; frags preloaded to hide LDS lat ----
__device__ __forceinline__ void mma_chunk(const u16* __restrict__ sAb,
                                          const u16* __restrict__ sBb,
                                          v8f acc[8], int w, int l15, bool hi) {
#pragma unroll
  for (int dk = 0; dk < 2; ++dk) {
    const int kb = dk * 32;
    Frag a;
    const int arow = w * 16 + l15, ac = kb + (hi ? 8 : 0);
    a.u[0] = *(const uint4*)&sAb[arow * SAS + ac];
    a.u[1] = *(const uint4*)&sAb[arow * SAS + ac + 16];
    Frag bf[8];
#pragma unroll
    for (int t = 0; t < 8; ++t) {
      const int brow = t * 16 + l15, bc = kb + (hi ? 8 : 0);
      bf[t].u[0] = *(const uint4*)&sBb[brow * SAS + bc];
      bf[t].u[1] = *(const uint4*)&sBb[brow * SAS + bc + 16];
    }
#pragma unroll
    for (int t = 0; t < 8; ++t)
      acc[t] = __builtin_amdgcn_wmma_f32_16x16x32_bf16(
          false, a.v, false, bf[t].v, (short)0, acc[t], false, false);
  }
}

// ----- the S-tile GEMM: acc[8] += (Arows x Brows^T) over d, LDS staged -----
__device__ __forceinline__ void stile_gemm(u16* __restrict__ sA, u16* __restrict__ sB,
                                           const u16* __restrict__ gA,
                                           const u16* __restrict__ gB,
                                           v8f acc[8], int tid, int w, int l15, bool hi) {
#if USE_TDM
  // double-buffered TDM pipeline: wave0 issues DMA, all waves compute
  if (w == 0 && (tid & 31) == 0) {
    tdm_load_tile(gA, (u32)(uintptr_t)&sA[0]);
    tdm_load_tile(gB, (u32)(uintptr_t)&sB[0]);
  }
#pragma unroll
  for (int dc = 0; dc < NCH; ++dc) {
    const int buf = dc & 1;
    if (w == 0 && (tid & 31) == 0) {
      if (dc + 1 < NCH) {
        const int nbuf = (dc + 1) & 1;
        tdm_load_tile(gA + (dc + 1) * DC, (u32)(uintptr_t)&sA[nbuf * TILE * SAS]);
        tdm_load_tile(gB + (dc + 1) * DC, (u32)(uintptr_t)&sB[nbuf * TILE * SAS]);
        __builtin_amdgcn_s_wait_tensorcnt(2);   // chunk dc landed, dc+1 in flight
      } else {
        __builtin_amdgcn_s_wait_tensorcnt(0);
      }
    }
    __syncthreads();                             // chunk dc visible to all waves
    mma_chunk(&sA[buf * TILE * SAS], &sB[buf * TILE * SAS], acc, w, l15, hi);
    __syncthreads();                             // buf free before TDM reuses it
  }
#else
#pragma unroll
  for (int dc = 0; dc < NCH; ++dc) {
    load_chunk(sA, sB, gA, gB, dc, tid);
    __syncthreads();
    mma_chunk(sA, sB, acc, w, l15, hi);
    __syncthreads();
  }
#endif
}

#if USE_TDM
#define LDS_BUFS 2
#else
#define LDS_BUFS 1
#endif

// ---------------- K2: row pass -> lse[b,q] (online softmax) ----------------
__global__ __launch_bounds__(256)
void attn_rowpass(const u16* __restrict__ Qb, const u16* __restrict__ Kb,
                  float* __restrict__ lse) {
  __shared__ __align__(16) u16 sA[LDS_BUFS * TILE * SAS];
  __shared__ __align__(16) u16 sB[LDS_BUFS * TILE * SAS];
  const int b = blockIdx.x >> 4, qt = blockIdx.x & 15;
  const int qbase = qt * TILE;
  const int tid = threadIdx.x, lane = tid & 31, w = tid >> 5;
  const int l15 = lane & 15; const bool hi = lane >= 16;

  const u16* Qrow = Qb + ((size_t)b * QQ + qbase) * DD;
  const u16* K0   = Kb + (size_t)b * KK * DD;

  float m[8], s[8];
#pragma unroll
  for (int r = 0; r < 8; ++r) { m[r] = NEGMAX; s[r] = 0.f; }

  for (int kt = 0; kt < KK / TILE; ++kt) {
    v8f acc[8];
    const v8f zro = {0.f, 0.f, 0.f, 0.f, 0.f, 0.f, 0.f, 0.f};
#pragma unroll
    for (int t = 0; t < 8; ++t) acc[t] = zro;

    stile_gemm(sA, sB, Qrow, K0 + (size_t)kt * TILE * DD, acc, tid, w, l15, hi);

    // online row max/sum update (rows r / r+8 by lane half; cols on 16 lanes)
#pragma unroll
    for (int r = 0; r < 8; ++r) {
      float x[8], tmax = NEGMAX;
#pragma unroll
      for (int t = 0; t < 8; ++t) { x[t] = acc[t][r] * SCALE; tmax = fmaxf(tmax, x[t]); }
      const float nm = fmaxf(m[r], tmax);
      s[r] *= __expf(m[r] - nm);
#pragma unroll
      for (int t = 0; t < 8; ++t) s[r] += __expf(x[t] - nm);
      m[r] = nm;
    }
    __syncthreads();   // all reads of acc done before LDS is restaged
  }
  // 16-lane combine of (max, sum), then write lse
#pragma unroll
  for (int r = 0; r < 8; ++r) {
    float mv = m[r], sv = s[r];
#pragma unroll
    for (int mask = 1; mask < 16; mask <<= 1) {
      const float om = __shfl_xor(mv, mask, 16);
      const float os = __shfl_xor(sv, mask, 16);
      const float nm = fmaxf(mv, om);
      sv = sv * __expf(mv - nm) + os * __expf(om - nm);
      mv = nm;
    }
    if (l15 == 0) {
      const int q = qbase + w * 16 + r + (hi ? 8 : 0);
      lse[b * QQ + q] = mv + __logf(sv);
    }
  }
}

// ------- K3: column pass -> colsum[b,k] = sum_q alpha, argmax_q alpha -------
__global__ __launch_bounds__(256)
void attn_colpass(const u16* __restrict__ Kb, const u16* __restrict__ Qb,
                  const float* __restrict__ lse, float* __restrict__ colsum,
                  int* __restrict__ aggr) {
  __shared__ __align__(16) u16 sA[LDS_BUFS * TILE * SAS];
  __shared__ __align__(16) u16 sB[LDS_BUFS * TILE * SAS];
  __shared__ float sLse[TILE];
  const int b = blockIdx.x >> 4, kt = blockIdx.x & 15;
  const int kbase = kt * TILE;
  const int tid = threadIdx.x, lane = tid & 31, w = tid >> 5;
  const int l15 = lane & 15; const bool hi = lane >= 16;

  const u16* Krow = Kb + ((size_t)b * KK + kbase) * DD;
  const u16* Q0   = Qb + (size_t)b * QQ * DD;

  float bestV[8], sum[8]; int bestI[8];
#pragma unroll
  for (int r = 0; r < 8; ++r) { bestV[r] = NEGMAX; sum[r] = 0.f; bestI[r] = 0; }

  for (int qt = 0; qt < QQ / TILE; ++qt) {
    if (tid < TILE) sLse[tid] = lse[b * QQ + qt * TILE + tid];
    v8f acc[8];
    const v8f zro = {0.f, 0.f, 0.f, 0.f, 0.f, 0.f, 0.f, 0.f};
#pragma unroll
    for (int t = 0; t < 8; ++t) acc[t] = zro;

    // A = K rows, B = Q rows  (S' = K Q^T)
    stile_gemm(sA, sB, Krow, Q0 + (size_t)qt * TILE * DD, acc, tid, w, l15, hi);

    // per-column stats: x = s - lse[q]; argmax (first index) + sum of exp
#pragma unroll
    for (int t = 0; t < 8; ++t) {
      const float lv = sLse[t * 16 + l15];
      const int qcol = qt * TILE + t * 16 + l15;
#pragma unroll
      for (int r = 0; r < 8; ++r) {
        const float x = acc[t][r] * SCALE - lv;
        sum[r] += __expf(x);
        if (x > bestV[r]) { bestV[r] = x; bestI[r] = qcol; }
      }
    }
    __syncthreads();   // protect sLse + LDS bufs before next iteration
  }
  // 16-lane combine: sum-add + argmax with lowest-index tie break
#pragma unroll
  for (int r = 0; r < 8; ++r) {
    float v = bestV[r], sv = sum[r]; int idx = bestI[r];
#pragma unroll
    for (int mask = 1; mask < 16; mask <<= 1) {
      const float ov = __shfl_xor(v, mask, 16);
      const int   oi = __shfl_xor(idx, mask, 16);
      const float os = __shfl_xor(sv, mask, 16);
      sv += os;
      if (ov > v || (ov == v && oi < idx)) { v = ov; idx = oi; }
    }
    if (l15 == 0) {
      const int k = kbase + w * 16 + r + (hi ? 8 : 0);
      colsum[b * KK + k] = sv;
      aggr[b * KK + k] = idx;
    }
  }
}

// ---------------- K4: out[b,:] = colsum^T * V (memory-bound GEMV) ----------
__global__ __launch_bounds__(512)
void attn_outpass(const float* __restrict__ V, const float* __restrict__ colsum,
                  float* __restrict__ out) {
  __shared__ float sC[KK];
  const int b = blockIdx.x, tid = threadIdx.x;
  for (int i = tid; i < KK; i += 512) sC[i] = colsum[b * KK + i];
  __syncthreads();
  const float* Vb = V + (size_t)b * KK * DD + tid;
  float acc = 0.f;
#pragma unroll 8
  for (int k = 0; k < KK; ++k) acc += sC[k] * Vb[(size_t)k * DD];
  out[b * DD + tid] = acc;
}

extern "C" void kernel_launch(void* const* d_in, const int* in_sizes, int n_in,
                              void* d_out, int out_size, void* d_ws, size_t ws_size,
                              hipStream_t stream) {
  const float* queries = (const float*)d_in[0];
  const float* keys    = (const float*)d_in[1];
  const float* values  = (const float*)d_in[2];

  // workspace layout
  u16*  Qb = (u16*)d_ws;
  u16*  Kb = Qb + (size_t)BB * QQ * DD;
  float* lse    = (float*)(Kb + (size_t)BB * KK * DD);
  float* colsum = lse + (size_t)BB * QQ;

  float* out  = (float*)d_out;                   // [B, DD] f32
  int*   aggr = (int*)((float*)d_out + BB * DD); // [B, KK] int32

  const int n4 = (BB * QQ * DD) / 4;
  attn_cvt_bf16<<<4096, 256, 0, stream>>>(queries, Qb, n4);
  attn_cvt_bf16<<<4096, 256, 0, stream>>>(keys, Kb, n4);
  attn_rowpass<<<BB * (QQ / TILE), 256, 0, stream>>>(Qb, Kb, lse);
  attn_colpass<<<BB * (KK / TILE), 256, 0, stream>>>(Kb, Qb, lse, colsum, aggr);
  attn_outpass<<<BB, 512, 0, stream>>>(values, colsum, out);
}
